// BuildNN_GNN_MTL_57131654971963
// MI455X (gfx1250) — compile-verified, matrix operand order checked by reading, not run on hardware
//
#include <hip/hip_runtime.h>
#include <hip/hip_bf16.h>

// ---------------------------------------------------------------------------
// BuildNN GNN MTL forward for MI455X (gfx1250, wave32, WMMA bf16)
//
// Dominant work: 3 CGConv layers = fused gather + dual GEMM
// (800k x 384 @ 384 x 128, gate & softplus) on v_wmma_f32_16x16x32_bf16.
// Node features mirrored in bf16 (12.8 MB -> L2 resident) so edge gathers hit
// L2. Edge features recomputed per tile from the raw 8-dim edge_attr.
// Register blocking: 4 edge tiles (64 edges) per block so each weight fragment
// load feeds 8 WMMAs (4x less L2 weight traffic than 1 tile/block).
// Scatter via global_atomic_add_f32 into f32 accumulator.
// ---------------------------------------------------------------------------

#define NNEU 128          // node / edge neuron width
#define KTOT 384          // 2*NN + NE
#define KSTEPS 12         // KTOT / 32
#define MTILES 4          // 16-edge tiles per block (64 edges)
#define WFRAG_ELEMS (2 * KSTEPS * 8 * 512)   // 2 mats x 12 steps x 8 waves x 32 lanes x 16

typedef __attribute__((ext_vector_type(16))) __bf16 v16bf;
typedef __attribute__((ext_vector_type(8)))  float  v8f;

static __device__ __forceinline__ unsigned short f32_to_bf16(float f) {
    unsigned int u = __float_as_uint(f);
    unsigned int lsb = (u >> 16) & 1u;
    u += 0x7fffu + lsb;                   // round-to-nearest-even
    return (unsigned short)(u >> 16);
}

// (kk in 0..31) -> fragment (lane_half, elem) per ISA 16-bit A-matrix 16x32 table
static __device__ __forceinline__ void kk_to_frag(int kk, int& lane_half, int& elem) {
    lane_half = (kk >> 3) & 1;
    elem = ((kk >> 4) << 3) | (kk & 7);
}

// ---------------------------------------------------------------------------
__global__ void fillzero_kernel(float* p, int n) {
    int g = blockIdx.x * blockDim.x + threadIdx.x;
    if (g < n) p[g] = 0.0f;
}

// x[N,32] @ W[32,128] + b -> xf (f32) and xh (bf16 mirror)
__global__ __launch_bounds__(256) void node_expand_kernel(
    const float* __restrict__ x, const float* __restrict__ w,
    const float* __restrict__ b, float* __restrict__ xf,
    unsigned short* __restrict__ xh, int N) {
    int g = blockIdx.x * 256 + threadIdx.x;
    if (g >= N * NNEU) return;
    int n = g >> 7, c = g & 127;
    const float* xr = x + (size_t)n * 32;
    float acc = b[c];
#pragma unroll
    for (int k = 0; k < 32; ++k) acc += xr[k] * w[k * NNEU + c];
    xf[g] = acc;
    xh[g] = f32_to_bf16(acc);
}

// Pre-swizzle conv weights (f and s, each [384,128] f32) into bf16 WMMA
// B-fragment layout: [mat][step][wave][lane][elem]
__global__ __launch_bounds__(256) void wprep_kernel(
    const float* __restrict__ fw, const float* __restrict__ sw,
    unsigned short* __restrict__ wfrag) {
    int g = blockIdx.x * 256 + threadIdx.x;
    if (g >= WFRAG_ELEMS) return;
    int elem = g & 15;
    int lane = (g >> 4) & 31;
    int wave = (g >> 9) & 7;
    int step = (g >> 12) % KSTEPS;
    int mat  = g / (KSTEPS * 8 * 512);
    int col  = wave * 16 + (lane & 15);
    int half = lane >> 4;
    int kk   = (((elem >> 3) & 1) << 4) | (half << 3) | (elem & 7);
    int K    = step * 32 + kk;
    const float* w = mat ? sw : fw;
    wfrag[g] = f32_to_bf16(w[(size_t)K * NNEU + col]);
}

// Fused CGConv edge kernel: gather -> dual WMMA GEMM (4 edge tiles register-
// blocked against shared weight fragments) -> sigmoid*softplus -> atomic
// scatter-add into acc (pre-initialized with x for the residual).
__global__ __launch_bounds__(256) void conv_edge_wmma_kernel(
    const unsigned short* __restrict__ xh,   // [N,128] bf16
    const int* __restrict__ ei,              // [2,E] (row0=src, row1=dst)
    const float* __restrict__ edge_attr,     // [E,8]
    const unsigned short* __restrict__ wfrag,// pre-swizzled f/s weights
    const float* __restrict__ we_w,          // lin_edge w [8,128]
    const float* __restrict__ we_b,          // lin_edge b [128]
    const float* __restrict__ fb,            // conv f bias [128]
    const float* __restrict__ sb,            // conv s bias [128]
    float* __restrict__ acc,                 // [N,128] f32
    int E) {
    __shared__ unsigned short ldsA[MTILES * KSTEPS * 512];  // 48 KB, fragment layout
    __shared__ int ldsSrc[16 * MTILES];
    __shared__ int ldsDst[16 * MTILES];

    const int t = threadIdx.x;
    const int e0 = blockIdx.x * (16 * MTILES);

    if (t < 16 * MTILES) {
        int e = e0 + t;
        ldsSrc[t] = (e < E) ? ei[e] : -1;
        ldsDst[t] = (e < E) ? ei[E + e] : -1;
    }
    __syncthreads();

    // --- gather x[dst] (K 0..127), x[src] (K 128..255): 2048 pairs per tile ---
#pragma unroll
    for (int it = 0; it < 8 * MTILES; ++it) {
        int p = t + it * 256;
        int tile = p >> 11;        // 2048 pairs per 16-edge tile
        int q = p & 2047;
        int m = q >> 7;            // edge row within tile
        int j = q & 127;           // pair index, K = 2j
        int K = 2 * j;
        int row = tile * 16 + m;
        int idx = (K < 128) ? ldsDst[row] : ldsSrc[row];
        int koff = (K < 128) ? K : (K - 128);
        unsigned int val = 0;
        if (idx >= 0) val = *(const unsigned int*)(xh + (size_t)idx * NNEU + koff);
        int step = K >> 5, kk = K & 31, lh, el;
        kk_to_frag(kk, lh, el);
        *(unsigned int*)(&ldsA[(tile * KSTEPS + step) * 512 + (m + 16 * lh) * 16 + el]) = val;
    }

    // --- compute e = edge_attr @ We + be for K 256..383: 1024 pairs per tile ---
#pragma unroll
    for (int it = 0; it < 4 * MTILES; ++it) {
        int p = t + it * 256;
        int tile = p >> 10;        // 1024 pairs per 16-edge tile
        int q = p & 1023;
        int m = q >> 6;            // edge row within tile
        int j = q & 63;            // output col pair (2j, 2j+1)
        int e = e0 + tile * 16 + m;
        float a0 = 0.0f, a1 = 0.0f;
        if (e < E) {
            const float* er = edge_attr + (size_t)e * 8;
            int c0 = 2 * j;
            a0 = we_b[c0];
            a1 = we_b[c0 + 1];
#pragma unroll
            for (int k = 0; k < 8; ++k) {
                float av = er[k];
                a0 += av * we_w[k * NNEU + c0];
                a1 += av * we_w[k * NNEU + c0 + 1];
            }
        }
        int K = 256 + 2 * j;
        int step = K >> 5, kk = K & 31, lh, el;
        kk_to_frag(kk, lh, el);
        unsigned int packed = (unsigned int)f32_to_bf16(a0) |
                              ((unsigned int)f32_to_bf16(a1) << 16);
        *(unsigned int*)(&ldsA[(tile * KSTEPS + step) * 512 + (m + 16 * lh) * 16 + el]) = packed;
    }
    __syncthreads();

    // --- dual WMMA, register-blocked over MTILES edge tiles ---
    const int wave = t >> 5;
    const int lane = t & 31;
    v8f cf[MTILES], cs[MTILES];
#pragma unroll
    for (int m = 0; m < MTILES; ++m) {
        cf[m] = (v8f){0.f, 0.f, 0.f, 0.f, 0.f, 0.f, 0.f, 0.f};
        cs[m] = (v8f){0.f, 0.f, 0.f, 0.f, 0.f, 0.f, 0.f, 0.f};
    }
#pragma unroll
    for (int k = 0; k < KSTEPS; ++k) {
        v16bf bf_ = *(const v16bf*)(wfrag + ((size_t)(0 * KSTEPS + k) * 8 + wave) * 512 + lane * 16);
        v16bf bs_ = *(const v16bf*)(wfrag + ((size_t)(1 * KSTEPS + k) * 8 + wave) * 512 + lane * 16);
#pragma unroll
        for (int m = 0; m < MTILES; ++m) {
            v16bf a = *(const v16bf*)(&ldsA[(m * KSTEPS + k) * 512 + lane * 16]);
            cf[m] = __builtin_amdgcn_wmma_f32_16x16x32_bf16(false, a, false, bf_, (short)0, cf[m], false, false);
            cs[m] = __builtin_amdgcn_wmma_f32_16x16x32_bf16(false, a, false, bs_, (short)0, cs[m], false, false);
        }
    }

    // --- m = sigmoid(f + bf) * softplus(s + bs); scatter-add to acc[dst] ---
    const int col = wave * 16 + (lane & 15);
    const int rbase = 8 * (lane >> 4);
    const float biasf = fb[col];
    const float biass = sb[col];
#pragma unroll
    for (int m = 0; m < MTILES; ++m) {
#pragma unroll
        for (int v = 0; v < 8; ++v) {
            int row = m * 16 + rbase + v;
            int d = ldsDst[row];
            if (d >= 0) {
                float fv = cf[m][v] + biasf;
                float sv = cs[m][v] + biass;
                float sig = 1.0f / (1.0f + expf(-fv));
                float sp  = (sv > 15.0f) ? sv : log1pf(expf(sv));
                atomicAdd(acc + (size_t)d * NNEU + col, sig * sp);
            }
        }
    }
}

// per-channel sum / sumsq over N nodes -> stats[0..127]=sum, [128..255]=sumsq
__global__ __launch_bounds__(256) void bn_stats_kernel(
    const float* __restrict__ acc, float* __restrict__ stats, int N) {
    int g = blockIdx.x * 256 + threadIdx.x;
    int c = g & 127;
    int r0 = g >> 7;
    int rstride = (gridDim.x * 256) >> 7;
    float s = 0.0f, q = 0.0f;
    for (int n = r0; n < N; n += rstride) {
        float v = acc[(size_t)n * NNEU + c];
        s += v;
        q += v * v;
    }
    atomicAdd(stats + c, s);
    atomicAdd(stats + 128 + c, q);
}

// x = relu(bn(acc)); writes f32 + bf16 mirror
__global__ __launch_bounds__(256) void bn_apply_kernel(
    const float* __restrict__ acc, const float* __restrict__ stats,
    const float* __restrict__ gam, const float* __restrict__ bet,
    float* __restrict__ xf, unsigned short* __restrict__ xh, int N) {
    int g = blockIdx.x * 256 + threadIdx.x;
    if (g >= N * NNEU) return;
    int c = g & 127;
    float inv = 1.0f / (float)N;
    float mean = stats[c] * inv;
    float var = stats[128 + c] * inv - mean * mean;
    float scale = gam[c] * rsqrtf(var + 1e-5f);
    float shift = bet[c] - mean * scale;
    float v = acc[g] * scale + shift;
    v = v > 0.0f ? v : 0.0f;
    xf[g] = v;
    xh[g] = f32_to_bf16(v);
}

// global_add_pool: h[batch[n], c] += x[n, c]   (h row stride = 134)
__global__ __launch_bounds__(256) void pool_scatter_kernel(
    const float* __restrict__ xf, const int* __restrict__ batch,
    float* __restrict__ h, int N) {
    int g = blockIdx.x * 256 + threadIdx.x;
    if (g >= N * NNEU) return;
    int n = g >> 7, c = g & 127;
    atomicAdd(h + (size_t)batch[n] * 134 + c, xf[g]);
}

// h[b, 128+j] = (gf[b,j]-gm[j])/(gs[j]+1e-8)
__global__ void globfeat_kernel(const float* __restrict__ gf,
                                const float* __restrict__ gm,
                                const float* __restrict__ gs,
                                float* __restrict__ h, int B) {
    int g = blockIdx.x * blockDim.x + threadIdx.x;
    if (g >= B * 6) return;
    int b = g / 6, j = g % 6;
    h[(size_t)b * 134 + 128 + j] = (gf[g] - gm[j]) / (gs[j] + 1e-8f);
}

// generic dense: out[B,dout] = in[B,din] @ W[din,dout] + b
__global__ __launch_bounds__(256) void lin_kernel(
    const float* __restrict__ in, const float* __restrict__ w,
    const float* __restrict__ b, float* __restrict__ out,
    int B, int din, int dout) {
    int g = blockIdx.x * 256 + threadIdx.x;
    if (g >= B * dout) return;
    int r = g / dout, c = g % dout;
    const float* ir = in + (size_t)r * din;
    float acc = b[c];
    for (int k = 0; k < din; ++k) acc += ir[k] * w[(size_t)k * dout + c];
    out[g] = acc;
}

// BN(training stats over B rows) + relu, in place. grid = C blocks.
__global__ __launch_bounds__(256) void bnrelu_kernel(
    float* __restrict__ h, const float* __restrict__ gam,
    const float* __restrict__ bet, int B, int C) {
    int c = blockIdx.x;
    __shared__ float rs[256], rq[256];
    float s = 0.0f, q = 0.0f;
    for (int r = threadIdx.x; r < B; r += 256) {
        float v = h[(size_t)r * C + c];
        s += v;
        q += v * v;
    }
    rs[threadIdx.x] = s;
    rq[threadIdx.x] = q;
    __syncthreads();
    for (int o = 128; o > 0; o >>= 1) {
        if (threadIdx.x < o) {
            rs[threadIdx.x] += rs[threadIdx.x + o];
            rq[threadIdx.x] += rq[threadIdx.x + o];
        }
        __syncthreads();
    }
    float mean = rs[0] / (float)B;
    float var = rq[0] / (float)B - mean * mean;
    float scale = gam[c] * rsqrtf(var + 1e-5f);
    float shift = bet[c] - mean * scale;
    for (int r = threadIdx.x; r < B; r += 256) {
        size_t i = (size_t)r * C + c;
        float v = h[i] * scale + shift;
        h[i] = v > 0.0f ? v : 0.0f;
    }
}

// sigmoid head: out[b] = sigmoid(y[b,:] . w + b0)
__global__ __launch_bounds__(256) void sighead_kernel(
    const float* __restrict__ y, const float* __restrict__ w,
    const float* __restrict__ b0, float* __restrict__ out, int B, int din) {
    int r = blockIdx.x * 256 + threadIdx.x;
    if (r >= B) return;
    const float* yr = y + (size_t)r * din;
    float acc = b0[0];
    for (int k = 0; k < din; ++k) acc += yr[k] * w[k];
    out[r] = 1.0f / (1.0f + expf(-acc));
}

// ---------------------------------------------------------------------------
extern "C" void kernel_launch(void* const* d_in, const int* in_sizes, int n_in,
                              void* d_out, int out_size, void* d_ws, size_t ws_size,
                              hipStream_t stream) {
    (void)n_in; (void)out_size; (void)ws_size;
    const float* x    = (const float*)d_in[0];
    const int*   ei   = (const int*)d_in[1];
    const float* ea   = (const float*)d_in[2];
    const int*   batch= (const int*)d_in[3];
    const float* gf   = (const float*)d_in[4];
    const float* gm   = (const float*)d_in[5];
    const float* gs   = (const float*)d_in[6];
    const int N = in_sizes[0] / 32;
    const int E = in_sizes[1] / 2;
    const int B = in_sizes[4] / 6;

    // --- params: JAX pytree order (sorted dict keys at each level) ---
    int pi = 7;
    const float *conv_bnb[3], *conv_bng[3], *conv_fb[3], *conv_fw[3], *conv_sb[3], *conv_sw[3];
    for (int l = 0; l < 3; ++l) {               // convs: bn_b, bn_g, f.b, f.w, s.b, s.w
        conv_bnb[l] = (const float*)d_in[pi++];
        conv_bng[l] = (const float*)d_in[pi++];
        conv_fb[l]  = (const float*)d_in[pi++];
        conv_fw[l]  = (const float*)d_in[pi++];
        conv_sb[l]  = (const float*)d_in[pi++];
        conv_sw[l]  = (const float*)d_in[pi++];
    }
    const float *hb[5][2], *hbnb[5][2], *hbng[5][2], *hw[5][2], *hsb[5], *hsw[5];
    for (int h = 0; h < 5; ++h) {               // heads: layers[{b,bn_b,bn_g,w}x2], sig{b,w}
        for (int l = 0; l < 2; ++l) {
            hb[h][l]   = (const float*)d_in[pi++];
            hbnb[h][l] = (const float*)d_in[pi++];
            hbng[h][l] = (const float*)d_in[pi++];
            hw[h][l]   = (const float*)d_in[pi++];
        }
        hsb[h] = (const float*)d_in[pi++];
        hsw[h] = (const float*)d_in[pi++];
    }
    const float* le_b = (const float*)d_in[pi++];   // lin_edge: b, w
    const float* le_w = (const float*)d_in[pi++];
    const float* ln_b = (const float*)d_in[pi++];   // lin_node: b, w
    const float* ln_w = (const float*)d_in[pi++];
    const float *sh_b[3], *sh_bnb[3], *sh_bng[3], *sh_w[3];
    for (int l = 0; l < 3; ++l) {               // shared: b, bn_b, bn_g, w
        sh_b[l]   = (const float*)d_in[pi++];
        sh_bnb[l] = (const float*)d_in[pi++];
        sh_bng[l] = (const float*)d_in[pi++];
        sh_w[l]   = (const float*)d_in[pi++];
    }

    // --- workspace carve ---
    char* ws = (char*)d_ws;
    size_t off = 0;
    auto carve = [&](size_t bytes) -> void* {
        void* p = ws + off;
        off = (off + bytes + 255) & ~(size_t)255;
        return p;
    };
    float*          xf    = (float*)carve((size_t)N * NNEU * 4);
    unsigned short* xh    = (unsigned short*)carve((size_t)N * NNEU * 2);
    float*          acc   = (float*)carve((size_t)N * NNEU * 4);
    unsigned short* wfrag = (unsigned short*)carve((size_t)WFRAG_ELEMS * 2);
    float*          stats = (float*)carve(256 * 4);
    float*          hA    = (float*)carve((size_t)B * 512 * 4);
    float*          hB    = (float*)carve((size_t)B * 512 * 4);
    float*          hC    = (float*)carve((size_t)B * 128 * 4);
    float*          hD    = (float*)carve((size_t)B * 64 * 4);
    float*          dout  = (float*)d_out;

    const int blocks_conv = (E + 16 * MTILES - 1) / (16 * MTILES);

    // 1. node feature expansion (f32 + bf16 mirror)
    node_expand_kernel<<<(N * NNEU + 255) / 256, 256, 0, stream>>>(x, ln_w, ln_b, xf, xh, N);

    // 2. CGConv layers
    for (int l = 0; l < 3; ++l) {
        wprep_kernel<<<(WFRAG_ELEMS + 255) / 256, 256, 0, stream>>>(conv_fw[l], conv_sw[l], wfrag);
        hipMemcpyAsync(acc, xf, (size_t)N * NNEU * 4, hipMemcpyDeviceToDevice, stream);
        fillzero_kernel<<<1, 256, 0, stream>>>(stats, 256);
        conv_edge_wmma_kernel<<<blocks_conv, 256, 0, stream>>>(
            xh, ei, ea, wfrag, le_w, le_b, conv_fb[l], conv_sb[l], acc, E);
        bn_stats_kernel<<<256, 256, 0, stream>>>(acc, stats, N);
        bn_apply_kernel<<<(N * NNEU + 255) / 256, 256, 0, stream>>>(
            acc, stats, conv_bng[l], conv_bnb[l], xf, xh, N);
    }

    // 3. global_add_pool + global feature normalization into hA [B, 134]
    fillzero_kernel<<<(B * 134 + 255) / 256, 256, 0, stream>>>(hA, B * 134);
    pool_scatter_kernel<<<(N * NNEU + 255) / 256, 256, 0, stream>>>(xf, batch, hA, N);
    globfeat_kernel<<<(B * 6 + 255) / 256, 256, 0, stream>>>(gf, gm, gs, hA, B);

    // 4. shared MLP: 134 -> 512 -> 256 -> 128 (linear + BN + relu)
    lin_kernel<<<(B * 512 + 255) / 256, 256, 0, stream>>>(hA, sh_w[0], sh_b[0], hB, B, 134, 512);
    bnrelu_kernel<<<512, 256, 0, stream>>>(hB, sh_bng[0], sh_bnb[0], B, 512);
    lin_kernel<<<(B * 256 + 255) / 256, 256, 0, stream>>>(hB, sh_w[1], sh_b[1], hA, B, 512, 256);
    bnrelu_kernel<<<256, 256, 0, stream>>>(hA, sh_bng[1], sh_bnb[1], B, 256);
    lin_kernel<<<(B * 128 + 255) / 256, 256, 0, stream>>>(hA, sh_w[2], sh_b[2], hB, B, 256, 128);
    bnrelu_kernel<<<128, 256, 0, stream>>>(hB, sh_bng[2], sh_bnb[2], B, 128);

    // 5. five task heads: 128 -> 128 -> 64 -> sigmoid(1)
    for (int h = 0; h < 5; ++h) {
        lin_kernel<<<(B * 128 + 255) / 256, 256, 0, stream>>>(hB, hw[h][0], hb[h][0], hC, B, 128, 128);
        bnrelu_kernel<<<128, 256, 0, stream>>>(hC, hbng[h][0], hbnb[h][0], B, 128);
        lin_kernel<<<(B * 64 + 255) / 256, 256, 0, stream>>>(hC, hw[h][1], hb[h][1], hD, B, 128, 64);
        bnrelu_kernel<<<64, 256, 0, stream>>>(hD, hbng[h][1], hbnb[h][1], B, 64);
        sighead_kernel<<<(B + 255) / 256, 256, 0, stream>>>(hD, hsw[h], hsb[h], dout + (size_t)h * B, B, 64);
    }
}
